// RouterSpatialBlock_78804059947470
// MI455X (gfx1250) — compile-verified, matrix operand order checked by reading, not run on hardware
//
#include <hip/hip_runtime.h>
#include <hip/hip_bf16.h>
#include <math.h>

// ---------------------------------------------------------------------------
// RouterSpatialBlock forward for MI455X (gfx1250, wave32).
// All channel-mixing GEMMs (qkv / out-proj / pc conv1x1 / g_qkv / g_out) run
// through v_wmma_f32_16x16x32_f16 (f16 in, f32 accumulate). Weight tiles are
// staged into LDS with double-buffered global_load_async_to_lds_b128 (ASYNCcnt
// + s_wait_asynccnt), so the next tile's DMA overlaps the current WMMAs.
// Norms, softmax, linear-attention reductions and the mixer stay in fp32.
// ---------------------------------------------------------------------------

typedef __attribute__((ext_vector_type(16))) _Float16 v16h;
typedef __attribute__((ext_vector_type(8)))  _Float16 v8h;
typedef __attribute__((ext_vector_type(8)))  float    v8f;

#define B_     16
#define C_     256
#define H_     64
#define W_     64
#define HW_    4096          // H_*W_
#define NPIX   65536         // B_*HW_
#define GROUPS 32
#define CPG    8             // C_/GROUPS
#define INNER_ 64
#define HLOC   8
#define HD_    8             // INNER_/HLOC
#define HGLB   8
#define DGLB   32            // C_/HGLB
#define DESC_  128

static __device__ __forceinline__ float eluP1(float x) {
    // elu(x)+1 : x>0 -> x+1 ; x<=0 -> exp(x)
    return x > 0.f ? x + 1.f : expf(x);
}

// ---------------------------------------------------------------------------
// f32 -> f16 weight conversion (weights stay in [N,K] = [out,in] layout, which
// is exactly the B-operand layout our WMMA fragment loader wants).
// ---------------------------------------------------------------------------
__global__ void cvt_f16_kernel(const float* __restrict__ s, _Float16* __restrict__ d, int n) {
    int i = blockIdx.x * 256 + threadIdx.x;
    if (i < n) d[i] = (_Float16)s[i];
}

// ---------------------------------------------------------------------------
// GroupNorm statistics: one block per (b, group); reduce 8 channels * 4096.
// ---------------------------------------------------------------------------
__global__ void __launch_bounds__(256)
gn_stats_kernel(const float* __restrict__ X, float* __restrict__ mu,
                float* __restrict__ rstd, float eps) {
    int bg = blockIdx.x;                     // b*32 + g
    int b = bg >> 5, g = bg & 31;
    const float* base = X + ((size_t)b * C_ + (size_t)g * CPG) * HW_;
    float s = 0.f, s2 = 0.f;
    for (int i = threadIdx.x; i < CPG * HW_; i += 256) {
        float v = base[i];
        s += v; s2 += v * v;
    }
    __shared__ float r1[256], r2[256];
    r1[threadIdx.x] = s; r2[threadIdx.x] = s2;
    __syncthreads();
    for (int off = 128; off > 0; off >>= 1) {
        if (threadIdx.x < off) { r1[threadIdx.x] += r1[threadIdx.x + off];
                                 r2[threadIdx.x] += r2[threadIdx.x + off]; }
        __syncthreads();
    }
    if (threadIdx.x == 0) {
        float inv = 1.f / (CPG * HW_);
        float m = r1[0] * inv;
        float var = r2[0] * inv - m * m;
        mu[bg] = m;
        rstd[bg] = rsqrtf(var + eps);
    }
}

// ---------------------------------------------------------------------------
// Tokenize for shifted-window attention: edge-clamped gather from NCHW,
// group-norm, +pos, LayerNorm over C=256, emit f16 token rows [tok, 256].
// One block (256 threads == channels) per token.
// ---------------------------------------------------------------------------
__global__ void __launch_bounds__(256)
tokenize_ln_kernel(const float* __restrict__ X, const float* __restrict__ mu,
                   const float* __restrict__ rstd,
                   const float* __restrict__ gnw, const float* __restrict__ gnb,
                   const float* __restrict__ pos,
                   const float* __restrict__ lnw, const float* __restrict__ lnb,
                   _Float16* __restrict__ Atok, int G, int shift) {
    int tok = blockIdx.x;
    int c = threadIdx.x;
    int t = tok & 3;
    int tmp = tok >> 2;
    int gy = tmp % G; tmp /= G;
    int gx = tmp % G;
    int b  = tmp / G;
    int ph = gx * 2 + (t >> 1);
    int pw = gy * 2 + (t & 1);
    int h = ph - shift; h = h < 0 ? 0 : (h > H_ - 1 ? H_ - 1 : h);
    int w = pw - shift; w = w < 0 ? 0 : (w > W_ - 1 ? W_ - 1 : w);

    float v = X[(((size_t)b * C_ + c) * H_ + h) * W_ + w];
    int bg = b * GROUPS + (c >> 3);
    v = (v - mu[bg]) * rstd[bg] * gnw[c] + gnb[c] + pos[t * C_ + c];

    __shared__ float red[256];
    __shared__ float stat[2];
    red[c] = v; __syncthreads();
    for (int off = 128; off > 0; off >>= 1) {
        if (c < off) red[c] += red[c + off];
        __syncthreads();
    }
    if (c == 0) stat[0] = red[0] * (1.f / 256.f);
    __syncthreads();
    float d = v - stat[0];
    red[c] = d * d; __syncthreads();
    for (int off = 128; off > 0; off >>= 1) {
        if (c < off) red[c] += red[c + off];
        __syncthreads();
    }
    if (c == 0) stat[1] = red[0] * (1.f / 256.f);
    __syncthreads();
    float y = d * rsqrtf(stat[1] + 1e-5f) * lnw[c] + lnb[c];
    Atok[(size_t)tok * C_ + c] = (_Float16)y;
}

// ---------------------------------------------------------------------------
// Generic WMMA GEMM: C[M,N] = A[M,K](f16 row-major) * W[N,K](f16) + bias.
// Block = 8 waves (wave32): tile 128(M) x 64(N); each wave one 16x64 strip
// (4 accumulators reuse the A fragment). K % 32 == 0, N % 64 == 0.
//
// B (weight) tile for the whole block is staged into LDS by the async data
// mover (global_load_async_to_lds_b128), double-buffered: tile s+1 streams in
// while tile s feeds the WMMAs. LDS rows padded to 80B so the 16 rows of a
// fragment hit distinct banks.
//
// Fragment layout per ISA 7.12.2 (16-bit A 16x32): lane&15 = M (A) or N (B)
// row, half = lane>>4 selects K octets {0-7,16-23} (half 0) vs {8-15,24-31}.
//
// mode 0: outF[m*N+n] = acc+bias
// mode 1: f16 store with optional row scatter map (rowmap[m] < 0 -> drop)
// mode 2: fused residual-silu into NCHW f32: x = silu(x + (*scale)*(acc+bias))
//         (requires M == NPIX, N == C_)
// ---------------------------------------------------------------------------
#define BROW_STRIDE 40       // halves per LDS B-tile row (32 data + 8 pad = 80B)

__global__ void __launch_bounds__(256)
wmma_gemm_kernel(const _Float16* __restrict__ A, const _Float16* __restrict__ Wt,
                 const float* __restrict__ bias, int M, int N, int K,
                 int mode, const float* __restrict__ scale_ptr,
                 float* __restrict__ outF, _Float16* __restrict__ outH,
                 const int* __restrict__ rowmap, float* __restrict__ xio) {
    __shared__ __align__(16) _Float16 sB[2][64 * BROW_STRIDE];

    const int lane = threadIdx.x & 31;
    const int wave = threadIdx.x >> 5;
    const int m0 = blockIdx.y * 128 + wave * 16;
    const int n0 = blockIdx.x * 64;
    const int half = lane >> 4;       // K-octet selector
    const int lm = lane & 15;
    int mrow = m0 + lm;
    if (mrow > M - 1) mrow = M - 1;   // clamp tail rows; stores stay guarded
    const _Float16* Arow = A + (size_t)mrow * K;

    // Async B staging: thread -> (tile row, 16B part). 256 threads move the
    // full 64-row x 32-half (4KB) tile, 16B per lane per issue.
    const int brow  = threadIdx.x >> 2;      // 0..63  (n-local row)
    const int bpart = threadIdx.x & 3;       // 4 x 16B chunks per row
    const _Float16* gBrow = Wt + (size_t)(n0 + brow) * K + bpart * 8;
    unsigned ldsDst0 = (unsigned)(uintptr_t)&sB[0][brow * BROW_STRIDE + bpart * 8];
    unsigned ldsDst1 = (unsigned)(uintptr_t)&sB[1][brow * BROW_STRIDE + bpart * 8];

    const int nsteps = K / 32;
    {   // prologue: kick off tile 0
        const _Float16* src = gBrow;
        asm volatile("global_load_async_to_lds_b128 %0, %1, off"
                     :: "v"(ldsDst0), "v"(src) : "memory");
    }

    v8f acc[4] = {};
    int cur = 0;
    for (int s = 0; s < nsteps; ++s) {
        asm volatile("s_wait_asynccnt 0x0" ::: "memory");
        __syncthreads();                     // tile `cur` resident for all waves
        if (s + 1 < nsteps) {                // stream next tile into other buffer
            const _Float16* src = gBrow + (s + 1) * 32;
            unsigned dst = (cur == 0) ? ldsDst1 : ldsDst0;
            asm volatile("global_load_async_to_lds_b128 %0, %1, off"
                         :: "v"(dst), "v"(src) : "memory");
        }

        const int k0 = s * 32;
        const _Float16* ap = Arow + k0 + half * 8;
        v8h lo = *(const v8h*)ap;
        v8h hi = *(const v8h*)(ap + 16);
        v16h a;
#pragma unroll
        for (int i = 0; i < 8; ++i) { a[i] = lo[i]; a[i + 8] = hi[i]; }

#pragma unroll
        for (int t = 0; t < 4; ++t) {
            const _Float16* bp = &sB[cur][(t * 16 + lm) * BROW_STRIDE + half * 8];
            v8h blo = *(const v8h*)bp;
            v8h bhi = *(const v8h*)(bp + 16);
            v16h bf;
#pragma unroll
            for (int i = 0; i < 8; ++i) { bf[i] = blo[i]; bf[i + 8] = bhi[i]; }
            acc[t] = __builtin_amdgcn_wmma_f32_16x16x32_f16(
                false, a, false, bf, (short)0, acc[t], false, false);
        }
        cur ^= 1;
    }

    float sc = (mode == 2) ? *scale_ptr : 1.0f;
#pragma unroll
    for (int t = 0; t < 4; ++t) {
        int n = n0 + t * 16 + lm;
        float bv = bias[n];
#pragma unroll
        for (int r = 0; r < 8; ++r) {
            int m = m0 + r + half * 8;       // D layout: lanes 16-31 hold M=r+8
            if (m >= M) continue;
            float val = acc[t][r] + bv;
            if (mode == 0) {
                outF[(size_t)m * N + n] = val;
            } else if (mode == 1) {
                int rm = rowmap ? rowmap[m] : m;
                if (rm >= 0) outH[(size_t)rm * N + n] = (_Float16)val;
            } else {
                int b = m >> 12, hw = m & (HW_ - 1);
                size_t xi = ((size_t)(b * C_ + n)) * HW_ + hw;
                float v = xio[xi] + sc * val;
                xio[xi] = v / (1.f + expf(-v));   // silu
            }
        }
    }
}

// ---------------------------------------------------------------------------
// 2x2-window / 8-head attention over QKV f32 rows [tok,192]. One thread per
// (window, head): 4x4 scores, softmax, 4x8 output -> f16 rows [tok,64].
// ---------------------------------------------------------------------------
__global__ void win_attn_kernel(const float* __restrict__ QKV,
                                _Float16* __restrict__ OutA, int nWin) {
    int gid = blockIdx.x * 256 + threadIdx.x;
    if (gid >= nWin * HLOC) return;
    int hh = gid & 7;
    int win = gid >> 3;
    float q[4][HD_], k[4][HD_], v[4][HD_];
#pragma unroll
    for (int i = 0; i < 4; ++i) {
        const float* base = QKV + (size_t)(win * 4 + i) * (3 * INNER_) + hh * HD_;
#pragma unroll
        for (int d = 0; d < HD_; ++d) {
            q[i][d] = base[d];
            k[i][d] = base[INNER_ + d];
            v[i][d] = base[2 * INNER_ + d];
        }
    }
    const float rs = 0.35355339059327373f;   // 1/sqrt(8)
#pragma unroll
    for (int i = 0; i < 4; ++i) {
        float s[4], mx = -1e30f;
#pragma unroll
        for (int j = 0; j < 4; ++j) {
            float a = 0.f;
#pragma unroll
            for (int d = 0; d < HD_; ++d) a += q[i][d] * k[j][d];
            s[j] = a * rs;
            mx = fmaxf(mx, s[j]);
        }
        float den = 0.f;
#pragma unroll
        for (int j = 0; j < 4; ++j) { s[j] = expf(s[j] - mx); den += s[j]; }
        float inv = 1.f / den;
        _Float16* o = OutA + (size_t)(win * 4 + i) * INNER_ + hh * HD_;
#pragma unroll
        for (int d = 0; d < HD_; ++d) {
            float a = 0.f;
#pragma unroll
            for (int j = 0; j < 4; ++j) a += s[j] * v[j][d];
            o[d] = (_Float16)(a * inv);
        }
    }
}

// Row scatter map: token -> cropped pixel row (pixel-shuffle + crop), -1 drop.
__global__ void build_rowmap_kernel(int* __restrict__ rowmap, int G, int shift, int nTok) {
    int tok = blockIdx.x * 256 + threadIdx.x;
    if (tok >= nTok) return;
    int t = tok & 3;
    int tmp = tok >> 2;
    int gy = tmp % G; tmp /= G;
    int gx = tmp % G;
    int b  = tmp / G;
    int i = gx * 2 + (t >> 1) - shift;
    int j = gy * 2 + (t & 1) - shift;
    rowmap[tok] = (i >= 0 && i < H_ && j >= 0 && j < W_) ? ((b << 12) + i * W_ + j) : -1;
}

// GroupNorm -> f16 pixel-major rows [NPIX, 256] for the global-attn GEMMs.
__global__ void gn_rows_kernel(const float* __restrict__ X, const float* __restrict__ mu,
                               const float* __restrict__ rstd,
                               const float* __restrict__ gnw, const float* __restrict__ gnb,
                               _Float16* __restrict__ Rows) {
    size_t e = (size_t)blockIdx.x * 256 + threadIdx.x;  // NPIX*256 total
    int c = (int)(e & 255);
    size_t row = e >> 8;
    int b = (int)(row >> 12);
    int hw = (int)(row & (HW_ - 1));
    float v = X[((size_t)b * C_ + c) * HW_ + hw];
    int bg = b * GROUPS + (c >> 3);
    Rows[e] = (_Float16)((v - mu[bg]) * rstd[bg] * gnw[c] + gnb[c]);
}

// ---------------------------------------------------------------------------
// Linear attention: kv[d,e] = sum_n k'[n,d] v[n,e], ksum[d] = sum_n k'[n,d];
// one block per (b,h), LDS-staged in chunks of 64 pixels.
// ---------------------------------------------------------------------------
__global__ void __launch_bounds__(256)
la_kv_kernel(const _Float16* __restrict__ QKVg, float* __restrict__ kv,
             float* __restrict__ ksum) {
    int bh = blockIdx.x;                 // b*8 + h
    int b = bh >> 3, h = bh & 7;
    __shared__ float kk[64][DGLB + 1];
    __shared__ float vv[64][DGLB + 1];
    float accKV[4] = {0.f, 0.f, 0.f, 0.f};
    float accKS = 0.f;
    for (int n0 = 0; n0 < HW_; n0 += 64) {
        for (int idx = threadIdx.x; idx < 64 * DGLB; idx += 256) {
            int nn = idx >> 5, d = idx & 31;
            const _Float16* row = QKVg + ((size_t)(b * HW_ + n0 + nn)) * (3 * C_);
            kk[nn][d] = eluP1((float)row[C_ + h * DGLB + d]);
            vv[nn][d] = (float)row[2 * C_ + h * DGLB + d];
        }
        __syncthreads();
#pragma unroll
        for (int pp = 0; pp < 4; ++pp) {
            int p = threadIdx.x + pp * 256;
            int d = p >> 5, e = p & 31;
            float a = accKV[pp];
            for (int nn = 0; nn < 64; ++nn) a += kk[nn][d] * vv[nn][e];
            accKV[pp] = a;
        }
        if (threadIdx.x < DGLB) {
            float a = accKS;
            for (int nn = 0; nn < 64; ++nn) a += kk[nn][threadIdx.x];
            accKS = a;
        }
        __syncthreads();
    }
#pragma unroll
    for (int pp = 0; pp < 4; ++pp) {
        int p = threadIdx.x + pp * 256;
        kv[(size_t)bh * (DGLB * DGLB) + p] = accKV[pp];
    }
    if (threadIdx.x < DGLB) ksum[bh * DGLB + threadIdx.x] = accKS;
}

// out[n, h*32+e] = (q'[n]·kv[:,e]) / (q'[n]·ksum + 1e-6); one thread per (pix,h).
__global__ void la_out_kernel(const _Float16* __restrict__ QKVg,
                              const float* __restrict__ kv, const float* __restrict__ ksum,
                              _Float16* __restrict__ LA) {
    int gid = blockIdx.x * 256 + threadIdx.x;
    if (gid >= NPIX * HGLB) return;
    int h = gid & 7;
    size_t pix = (size_t)(gid >> 3);
    int b = (int)(pix >> 12);
    const _Float16* qrow = QKVg + pix * (3 * C_) + h * DGLB;
    float qv[DGLB];
#pragma unroll
    for (int d = 0; d < DGLB; ++d) qv[d] = eluP1((float)qrow[d]);
    const float* kvp = kv + (size_t)(b * HGLB + h) * (DGLB * DGLB);
    const float* ksp = ksum + (b * HGLB + h) * DGLB;
    float den = 1e-6f;
#pragma unroll
    for (int d = 0; d < DGLB; ++d) den += qv[d] * ksp[d];
    float inv = 1.f / den;
    for (int e = 0; e < DGLB; ++e) {
        float s = 0.f;
#pragma unroll
        for (int d = 0; d < DGLB; ++d) s += qv[d] * kvp[d * DGLB + e];
        LA[pix * C_ + h * DGLB + e] = (_Float16)(s * inv);
    }
}

// LA += depthwise3x3(v-image, zero-padded) + lepe_b; one thread per (pix, c).
__global__ void la_lepe_kernel(const _Float16* __restrict__ QKVg,
                               const float* __restrict__ lw, const float* __restrict__ lb,
                               _Float16* __restrict__ LA) {
    size_t e = (size_t)blockIdx.x * 256 + threadIdx.x;   // NPIX*256
    int c = (int)(e & 255);
    size_t pix = e >> 8;
    int b = (int)(pix >> 12);
    int hw = (int)(pix & (HW_ - 1));
    int i = hw >> 6, j = hw & 63;
    float acc = lb[c];
#pragma unroll
    for (int di = -1; di <= 1; ++di) {
#pragma unroll
        for (int dj = -1; dj <= 1; ++dj) {
            int ii = i + di, jj = j + dj;
            if (ii < 0 || ii >= H_ || jj < 0 || jj >= W_) continue;
            float v = (float)QKVg[((size_t)(b << 12) + ii * W_ + jj) * (3 * C_) + 2 * C_ + c];
            acc += lw[c * 9 + (di + 1) * 3 + (dj + 1)] * v;
        }
    }
    LA[e] = (_Float16)((float)LA[e] + acc);
}

// ---------------------------------------------------------------------------
// Residual mixer: per-(hist,b,c) sum / sumsq over HxW.
// ---------------------------------------------------------------------------
__global__ void __launch_bounds__(256)
mix_stats_kernel(const float* __restrict__ Xin, const float* __restrict__ X,
                 float* __restrict__ S1, float* __restrict__ S2) {
    int id = blockIdx.x;                  // hist*4096 + b*256 + c
    int hist = id >> 12;
    int bc = id & 4095;
    const float* base = (hist ? X : Xin) + (size_t)bc * HW_;
    float s = 0.f, s2 = 0.f;
    for (int i = threadIdx.x; i < HW_; i += 256) {
        float v = base[i]; s += v; s2 += v * v;
    }
    __shared__ float r1[256], r2[256];
    r1[threadIdx.x] = s; r2[threadIdx.x] = s2;
    __syncthreads();
    for (int off = 128; off > 0; off >>= 1) {
        if (threadIdx.x < off) { r1[threadIdx.x] += r1[threadIdx.x + off];
                                 r2[threadIdx.x] += r2[threadIdx.x + off]; }
        __syncthreads();
    }
    if (threadIdx.x == 0) { S1[id] = r1[0]; S2[id] = r2[0]; }
}

// GroupNorm-from-stats -> pooled descriptor -> key_w GEMV -> cosine vs query
// -> +recency*[-1,1] -> softmax(2). One block (128 threads) per batch.
__global__ void __launch_bounds__(128)
mix_weights_kernel(const float* __restrict__ S1, const float* __restrict__ S2,
                   const float* __restrict__ mgnw, const float* __restrict__ mgnb,
                   const float* __restrict__ keyw, const float* __restrict__ query,
                   const float* __restrict__ recency, float* __restrict__ wmix) {
    int b = blockIdx.x;
    int j = threadIdx.x;
    __shared__ float pooled[2][C_];
    __shared__ float gmean[2][GROUPS], grstd[2][GROUPS];
    __shared__ float red[128];

    if (j < 64) {
        int hist = j >> 5, g = j & 31;
        float s1 = 0.f, s2 = 0.f;
        for (int cc = 0; cc < CPG; ++cc) {
            int idx = hist * 4096 + b * C_ + g * CPG + cc;
            s1 += S1[idx]; s2 += S2[idx];
        }
        float inv = 1.f / (CPG * HW_);
        float mn = s1 * inv;
        float var = s2 * inv - mn * mn;
        gmean[hist][g] = mn;
        grstd[hist][g] = rsqrtf(var + 1e-5f);
    }
    __syncthreads();
    for (int c = j; c < C_; c += 128) {
        for (int hist = 0; hist < 2; ++hist) {
            int idx = hist * 4096 + b * C_ + c;
            pooled[hist][c] = mgnw[c] * (S1[idx] * (1.f / HW_) - gmean[hist][c >> 3])
                              * grstd[hist][c >> 3] + mgnb[c];
        }
    }
    __syncthreads();
    float d0 = 0.f, d1 = 0.f;
    for (int c = 0; c < C_; ++c) {
        float kw = keyw[j * C_ + c];
        d0 += pooled[0][c] * kw;
        d1 += pooled[1][c] * kw;
    }
    auto reduce = [&](float x) -> float {
        red[j] = x; __syncthreads();
        for (int off = 64; off > 0; off >>= 1) {
            if (j < off) red[j] += red[j + off];
            __syncthreads();
        }
        float r = red[0]; __syncthreads();
        return r;
    };
    float n0 = fmaxf(sqrtf(reduce(d0 * d0)), 1e-6f);
    float n1 = fmaxf(sqrtf(reduce(d1 * d1)), 1e-6f);
    float qv = query[j];
    float nq = fmaxf(sqrtf(reduce(qv * qv)), 1e-6f);
    float s0 = reduce((d0 / n0) * (qv / nq)) - (*recency);   // recency*(-1)
    float s1 = reduce((d1 / n1) * (qv / nq)) + (*recency);   // recency*(+1)
    if (j == 0) {
        float mx = fmaxf(s0, s1);
        float e0 = expf(s0 - mx), e1 = expf(s1 - mx);
        float inv = 1.f / (e0 + e1);
        wmix[b * 2 + 0] = e0 * inv;
        wmix[b * 2 + 1] = e1 * inv;
    }
}

__global__ void mix_apply_kernel(const float* __restrict__ X, const float* __restrict__ Xin,
                                 const float* __restrict__ wmix, const float* __restrict__ mixp,
                                 float* __restrict__ out) {
    size_t e = (size_t)blockIdx.x * 256 + threadIdx.x;     // NPIX*C_ total
    int b = (int)(e >> 20);                                // C_*HW_ = 2^20
    float xv = X[e], rv = Xin[e];
    float mixed = wmix[b * 2] * rv + wmix[b * 2 + 1] * xv;
    out[e] = xv + (*mixp) * (mixed - xv);
}

// ---------------------------------------------------------------------------
// Host-side orchestration.
// ---------------------------------------------------------------------------
static inline unsigned ceil_div_u(size_t a, size_t b) { return (unsigned)((a + b - 1) / b); }

extern "C" void kernel_launch(void* const* d_in, const int* in_sizes, int n_in,
                              void* d_out, int out_size, void* d_ws, size_t ws_size,
                              hipStream_t stream) {
    (void)in_sizes; (void)n_in; (void)out_size; (void)ws_size;

    // ---- inputs (setup_inputs dict order, depth-first) ----
    const float* x_in   = (const float*)d_in[0];
    const float* gn_w   = (const float*)d_in[1];
    const float* gn_b   = (const float*)d_in[2];
    // local[s] at base 3 + s*10:
    //  +0 pos, +1 ln_w, +2 ln_b, +3 qkv_w, +4 qkv_b, +5 out_w, +6 out_b,
    //  +7 pc_w, +8 pc_b, +9 scale
    const float* g_qkv_w = (const float*)d_in[23];
    const float* g_qkv_b = (const float*)d_in[24];
    const float* lepe_w  = (const float*)d_in[25];
    const float* lepe_b  = (const float*)d_in[26];
    const float* g_out_w = (const float*)d_in[27];
    const float* g_out_b = (const float*)d_in[28];
    const float* g_scale = (const float*)d_in[29];
    const float* m_gn_w  = (const float*)d_in[30];
    const float* m_gn_b  = (const float*)d_in[31];
    const float* key_w   = (const float*)d_in[32];
    const float* query   = (const float*)d_in[33];
    const float* recency = (const float*)d_in[34];
    const float* mixp    = (const float*)d_in[35];

    // ---- workspace bump allocator (~272 MB) ----
    char* base = (char*)d_ws;
    size_t off = 0;
    auto alloc = [&](size_t bytes) -> char* {
        char* p = base + off;
        off += (bytes + 255) & ~(size_t)255;
        return p;
    };
    const int TOKMAX = B_ * 33 * 33 * 4;                 // 69696 (shift=1)
    float*     X     = (float*)alloc((size_t)NPIX * C_ * 4);
    float*     MU    = (float*)alloc(B_ * GROUPS * 4);
    float*     RSTD  = (float*)alloc(B_ * GROUPS * 4);
    float*     S1    = (float*)alloc(2 * B_ * C_ * 4);
    float*     S2    = (float*)alloc(2 * B_ * C_ * 4);
    float*     WMIX  = (float*)alloc(B_ * 2 * 4);
    _Float16*  W16A  = (_Float16*)alloc((size_t)192 * 256 * 2);   // qkv_w
    _Float16*  W16B  = (_Float16*)alloc((size_t)256 * 64 * 2);    // out_w
    _Float16*  W16C  = (_Float16*)alloc((size_t)256 * 256 * 2);   // pc_w
    _Float16*  W16D  = (_Float16*)alloc((size_t)768 * 256 * 2);   // g_qkv_w
    _Float16*  W16E  = (_Float16*)alloc((size_t)256 * 256 * 2);   // g_out_w
    int*       RMAP  = (int*)alloc((size_t)TOKMAX * 4);
    _Float16*  ATOK  = (_Float16*)alloc((size_t)TOKMAX * 256 * 2);        // / Grows
    char*      QKVR  = alloc((size_t)NPIX * 768 * 2);                     // QKV f32 (Tok*192) / QKVg f16
    _Float16*  OUTA  = (_Float16*)alloc((size_t)NPIX * 256 * 2);          // OutA / LA rows
    char*      PREG  = alloc((size_t)NPIX * 256 * 2);                     // P f16 / kv+ksum

    float*    QKVf   = (float*)QKVR;          // [Tok, 192] f32 (local phase)
    _Float16* QKVg   = (_Float16*)QKVR;       // [NPIX, 768] f16 (global phase)
    _Float16* Pbuf   = (_Float16*)PREG;       // [NPIX, 256] f16
    float*    KVbuf  = (float*)PREG;          // [B*8, 32*32] f32
    float*    KSbuf  = KVbuf + (size_t)B_ * HGLB * DGLB * DGLB;

    // X <- input (inputs must not be mutated; X is updated in place).
    hipMemcpyAsync(X, x_in, (size_t)NPIX * C_ * 4, hipMemcpyDeviceToDevice, stream);

    auto cvt = [&](const float* s, _Float16* d, int n) {
        cvt_f16_kernel<<<ceil_div_u(n, 256), 256, 0, stream>>>(s, d, n);
    };
    auto gemm = [&](const _Float16* A, const _Float16* Wt, const float* bias,
                    int M, int N, int K, int mode, const float* sp,
                    float* oF, _Float16* oH, const int* rm, float* xio) {
        dim3 grid(N / 64, ceil_div_u(M, 128));
        wmma_gemm_kernel<<<grid, 256, 0, stream>>>(A, Wt, bias, M, N, K, mode, sp,
                                                   oF, oH, rm, xio);
    };

    // ================= local shifted-window blocks =================
    for (int s = 0; s < 2; ++s) {
        int pb = 3 + s * 10;
        const float* pos   = (const float*)d_in[pb + 0];
        const float* ln_w  = (const float*)d_in[pb + 1];
        const float* ln_b  = (const float*)d_in[pb + 2];
        const float* qkv_w = (const float*)d_in[pb + 3];
        const float* qkv_b = (const float*)d_in[pb + 4];
        const float* out_w = (const float*)d_in[pb + 5];
        const float* out_b = (const float*)d_in[pb + 6];
        const float* pc_w  = (const float*)d_in[pb + 7];
        const float* pc_b  = (const float*)d_in[pb + 8];
        const float* scale = (const float*)d_in[pb + 9];

        const int G = 32 + s;                    // (64 + 2*shift) / 2
        const int nWin = B_ * G * G;
        const int nTok = nWin * 4;

        cvt(qkv_w, W16A, 192 * 256);
        cvt(out_w, W16B, 256 * 64);
        cvt(pc_w,  W16C, 256 * 256);

        gn_stats_kernel<<<B_ * GROUPS, 256, 0, stream>>>(X, MU, RSTD, 1e-5f);
        tokenize_ln_kernel<<<nTok, 256, 0, stream>>>(X, MU, RSTD, gn_w, gn_b, pos,
                                                     ln_w, ln_b, ATOK, G, s);
        // qkv projection: [nTok,256] x [192,256]^T
        gemm(ATOK, W16A, qkv_b, nTok, 192, 256, 0, nullptr, QKVf, nullptr, nullptr, nullptr);
        win_attn_kernel<<<ceil_div_u((size_t)nWin * HLOC, 256), 256, 0, stream>>>(QKVf, OUTA, nWin);
        build_rowmap_kernel<<<ceil_div_u(nTok, 256), 256, 0, stream>>>(RMAP, G, s, nTok);
        // out projection, pixel-shuffle scatter into cropped pixel rows
        gemm(OUTA, W16B, out_b, nTok, 256, 64, 1, nullptr, nullptr, Pbuf, RMAP, nullptr);
        // pc conv1x1 + fused x = silu(x + scale*att)
        gemm(Pbuf, W16C, pc_b, NPIX, 256, 256, 2, scale, nullptr, nullptr, nullptr, X);
    }

    // ================= global linear attention =================
    cvt(g_qkv_w, W16D, 768 * 256);
    cvt(g_out_w, W16E, 256 * 256);

    gn_stats_kernel<<<B_ * GROUPS, 256, 0, stream>>>(X, MU, RSTD, 1e-5f);
    gn_rows_kernel<<<NPIX, 256, 0, stream>>>(X, MU, RSTD, gn_w, gn_b, ATOK);
    // g_qkv: [NPIX,256] x [768,256]^T -> f16 [NPIX,768]
    gemm(ATOK, W16D, g_qkv_b, NPIX, 768, 256, 1, nullptr, nullptr, QKVg, nullptr, nullptr);
    la_kv_kernel<<<B_ * HGLB, 256, 0, stream>>>(QKVg, KVbuf, KSbuf);
    la_out_kernel<<<ceil_div_u((size_t)NPIX * HGLB, 256), 256, 0, stream>>>(QKVg, KVbuf, KSbuf, OUTA);
    la_lepe_kernel<<<NPIX, 256, 0, stream>>>(QKVg, lepe_w, lepe_b, OUTA);
    // g_out + fused x = silu(x + g_scale*la)
    gemm(OUTA, W16E, g_out_b, NPIX, 256, 256, 2, g_scale, nullptr, nullptr, nullptr, X);

    // ================= residual mixer =================
    mix_stats_kernel<<<2 * B_ * C_, 256, 0, stream>>>(x_in, X, S1, S2);
    mix_weights_kernel<<<B_, 128, 0, stream>>>(S1, S2, m_gn_w, m_gn_b, key_w, query,
                                               recency, WMIX);
    mix_apply_kernel<<<NPIX, 256, 0, stream>>>(X, x_in, WMIX, mixp, (float*)d_out);
}
